// SE3Transformer_29618094473364
// MI455X (gfx1250) — matrix operand order repaired and checked by reading, not compiled, not run on hardware
//
#include <hip/hip_runtime.h>

// ---------------------------------------------------------------------------
// SE(3)-Transformer forward for MI455X (gfx1250, wave32).
// Dense contractions: generic WMMA GEMM (v_wmma_f32_16x16x32_f16, fp32 acc),
// 32x32 output per wave (4 WMMA tiles, A/B fragment reuse), fused row-gather
// for h[src]. B operands are pre-transposed + f16-converted + K-padded with
// zeros once per GEMM, so ALL K-loop loads are unconditional (B: contiguous
// global_load_b128; A: vector or clamped scalar loads -- out-of-K A values
// are finite duplicates multiplied by stored B zeros). No selects, no EXEC
// branching in the K-loop; only the 8-store epilogue is guarded.
// Segment softmax / aggregation use f32 global atomics (L2-resident).
// Edge intermediates are chunked (32K edges) to stay inside the 192MB L2.
// ---------------------------------------------------------------------------

#define NA_   20000
#define NR_   2000
#define EB_   80000
#define ER_   40000
#define EA_   480000
#define ECHUNK 32768
#define EPSV  1e-6f

typedef __attribute__((ext_vector_type(16))) _Float16 v16h;
typedef __attribute__((ext_vector_type(8)))  _Float16 v8h;
typedef __attribute__((ext_vector_type(8)))  float    v8f;

// ---------------- order-preserving float <-> uint for atomicMax ------------
__device__ __forceinline__ unsigned fenc(float f) {
  unsigned u = __float_as_uint(f);
  return (u >> 31) ? ~u : (u | 0x80000000u);
}
__device__ __forceinline__ float fdec(unsigned k) {
  return (k >> 31) ? __uint_as_float(k & 0x7fffffffu) : __uint_as_float(~k);
}

// ---- operand fragment loaders (ALL unconditional) -------------------------
// A fragment (f32 memory): lane holds row tile_m0+(lane&15); halves
// j=0..7 -> K=kA+j, j=8..15 -> K=16+kA+(j-8), kA = 8*(lane>=16).
__device__ __forceinline__ v16h loadA_vec(const float* __restrict__ Ar, int k0, int kA) {
  const float4* p0 = (const float4*)(Ar + k0 + kA);
  const float4* p1 = (const float4*)(Ar + k0 + 16 + kA);
  float4 x0 = p0[0], x1 = p0[1], x2 = p1[0], x3 = p1[1];
  v16h a;
  a[0]=(_Float16)x0.x;  a[1]=(_Float16)x0.y;  a[2]=(_Float16)x0.z;  a[3]=(_Float16)x0.w;
  a[4]=(_Float16)x1.x;  a[5]=(_Float16)x1.y;  a[6]=(_Float16)x1.z;  a[7]=(_Float16)x1.w;
  a[8]=(_Float16)x2.x;  a[9]=(_Float16)x2.y;  a[10]=(_Float16)x2.z; a[11]=(_Float16)x2.w;
  a[12]=(_Float16)x3.x; a[13]=(_Float16)x3.y; a[14]=(_Float16)x3.z; a[15]=(_Float16)x3.w;
  return a;
}
// scalar, unconditional (main loop, misaligned lda; k0+31 < K guaranteed)
__device__ __forceinline__ v16h loadA_scl(const float* __restrict__ Ar, int k0, int kA) {
  v16h a;
#pragma unroll
  for (int j = 0; j < 16; ++j) {
    int ka = k0 + ((j < 8) ? (kA + j) : (8 + kA + j));   // = 16+kA+(j-8)
    a[j] = (_Float16)Ar[ka];
  }
  return a;
}
// K-tail: clamp to K-1 (duplicate real value; B is zero-padded beyond K, so
// the product is exactly 0 and the duplicate is finite -> no NaN pollution)
__device__ __forceinline__ v16h loadA_clamp(const float* __restrict__ Ar, int k0, int kA,
                                            int K) {
  v16h a;
#pragma unroll
  for (int j = 0; j < 16; ++j) {
    int ka = k0 + ((j < 8) ? (kA + j) : (8 + kA + j));
    int kc = (ka < K) ? ka : (K - 1);
    a[j] = (_Float16)Ar[kc];
  }
  return a;
}
// B fragment (pre-transposed f16 [N][Kpad], zero-padded): contiguous loads.
__device__ __forceinline__ v16h loadBh(const _Float16* __restrict__ Bc, int ks) {
  const v8h* p = (const v8h*)(Bc + ks);
  v8h x0 = p[0], x1 = p[2];
  v16h b;
#pragma unroll
  for (int j = 0; j < 8; ++j) { b[j] = x0[j]; b[8 + j] = x1[j]; }
  return b;
}

__device__ __forceinline__ void store_tile(v8f acc, int mt, int col, bool cv, int hs,
    const float* __restrict__ bias, const float* __restrict__ Dadd, int ldd,
    float* __restrict__ C, int ldc, int M, int act)
{
  const int rbase = mt + hs * 8;    // C layout: vgpr i -> row rbase+i
#pragma unroll
  for (int i = 0; i < 8; ++i) {
    int r = rbase + i;
    if (r < M && cv) {
      float v = acc[i];
      if (bias) v += bias[col];
      if (Dadd) v += Dadd[(long)r * ldd + col];
      if (act == 1) v = fmaxf(v, 0.0f);
      else if (act == 2) v = (v > 0.0f) ? v : (__expf(v) - 1.0f);
      C[(long)r * ldc + col] = v;
    }
  }
}

// B [K,N] f32 row-major -> Bt [N,Kpad] f16, zero-padded in [K, Kpad)
__global__ void xposeh_kernel(const float* __restrict__ B, int K, int Kpad, int N,
                              _Float16* __restrict__ Bt)
{
  long t = (long)blockIdx.x * blockDim.x + threadIdx.x;
  if (t >= (long)N * Kpad) return;
  int n = (int)(t / Kpad), k = (int)(t % Kpad);
  Bt[t] = (k < K) ? (_Float16)B[(long)k * N + n] : (_Float16)0.0f;
}

// ---------------------------------------------------------------------------
// Generic WMMA GEMM:  C[M,N] = act(A[M,K] @ B[K,N] + bias + D)
//   - one wave per 32x32 output block (4 16x16 WMMA tiles, A/B reuse)
//   - optional rowidx gathers A rows (fused h[src] gather)
//   - B pre-transposed f16 [N][Kpad], zero-padded
// act: 0 = none, 1 = relu, 2 = elu
// ---------------------------------------------------------------------------
__global__ __launch_bounds__(32)
void gemm_wmma(const float* __restrict__ A, int lda, const int* __restrict__ rowidx,
               const _Float16* __restrict__ Bt, int Kpad,
               const float* __restrict__ bias,
               const float* __restrict__ Dadd, int ldd,
               float* __restrict__ C, int ldc,
               int M, int N, int K, int act)
{
  const int lane = threadIdx.x;
  const int hs   = lane >> 4;
  const int li   = lane & 15;
  const int m0   = blockIdx.y << 5;
  const int n0   = blockIdx.x << 5;

  // rows: clamp only (garbage rows are never stored)
  const int  row0 = m0 + li,       row1 = m0 + 16 + li;
  const int  rc0  = (row0 < M) ? row0 : (M - 1);
  const int  rc1  = (row1 < M) ? row1 : (M - 1);
  const long ar0  = rowidx ? (long)rowidx[rc0] : (long)rc0;
  const long ar1  = rowidx ? (long)rowidx[rc1] : (long)rc1;
  const float* __restrict__ Ar0 = A + ar0 * (long)lda;
  const float* __restrict__ Ar1 = A + ar1 * (long)lda;

  // cols: clamp only (invalid-column accumulators are never stored)
  const int  col0 = n0 + li,       col1 = n0 + 16 + li;
  const bool cv0  = col0 < N,      cv1  = col1 < N;
  const int  cc0  = cv0 ? col0 : (N - 1);
  const int  cc1  = cv1 ? col1 : (N - 1);
  const _Float16* __restrict__ Bc0 = Bt + (long)cc0 * Kpad;
  const _Float16* __restrict__ Bc1 = Bt + (long)cc1 * Kpad;

  const int kA = hs * 8;
  const int kB = hs * 8;

  const bool doM2 = (m0 + 16) < M;        // uniform
  const bool doN2 = (n0 + 16) < N;        // uniform
  const bool alA  = (lda & 3) == 0;       // uniform: vector vs scalar A loads
  const int  Kmain = K & ~31;

  v8f acc00 = {}, acc01 = {}, acc10 = {}, acc11 = {};

  for (int k0 = 0; k0 < Kmain; k0 += 32) {
    v16h a0 = alA ? loadA_vec(Ar0, k0, kA) : loadA_scl(Ar0, k0, kA);
    if (k0 + 32 < K) __builtin_prefetch(Ar0 + k0 + 32, 0, 0);   // global_prefetch_b8
    v16h b0 = loadBh(Bc0, k0 + kB);
    acc00 = __builtin_amdgcn_wmma_f32_16x16x32_f16(false, a0, false, b0,
                                                   (short)0, acc00, false, false);
    v16h b1 = {};
    if (doN2) {
      b1 = loadBh(Bc1, k0 + kB);
      acc01 = __builtin_amdgcn_wmma_f32_16x16x32_f16(false, a0, false, b1,
                                                     (short)0, acc01, false, false);
    }
    if (doM2) {
      v16h a1 = alA ? loadA_vec(Ar1, k0, kA) : loadA_scl(Ar1, k0, kA);
      acc10 = __builtin_amdgcn_wmma_f32_16x16x32_f16(false, a1, false, b0,
                                                     (short)0, acc10, false, false);
      if (doN2)
        acc11 = __builtin_amdgcn_wmma_f32_16x16x32_f16(false, a1, false, b1,
                                                       (short)0, acc11, false, false);
    }
  }
  if (Kmain < K) {      // K tail: clamped A (no selects), zero-padded B
    v16h a0 = loadA_clamp(Ar0, Kmain, kA, K);
    v16h b0 = loadBh(Bc0, Kmain + kB);
    acc00 = __builtin_amdgcn_wmma_f32_16x16x32_f16(false, a0, false, b0,
                                                   (short)0, acc00, false, false);
    v16h b1 = {};
    if (doN2) {
      b1 = loadBh(Bc1, Kmain + kB);
      acc01 = __builtin_amdgcn_wmma_f32_16x16x32_f16(false, a0, false, b1,
                                                     (short)0, acc01, false, false);
    }
    if (doM2) {
      v16h a1 = loadA_clamp(Ar1, Kmain, kA, K);
      acc10 = __builtin_amdgcn_wmma_f32_16x16x32_f16(false, a1, false, b0,
                                                     (short)0, acc10, false, false);
      if (doN2)
        acc11 = __builtin_amdgcn_wmma_f32_16x16x32_f16(false, a1, false, b1,
                                                       (short)0, acc11, false, false);
    }
  }

  store_tile(acc00, m0, col0, cv0, hs, bias, Dadd, ldd, C, ldc, M, act);
  if (doN2) store_tile(acc01, m0, col1, cv1, hs, bias, Dadd, ldd, C, ldc, M, act);
  if (doM2) {
    store_tile(acc10, m0 + 16, col0, cv0, hs, bias, Dadd, ldd, C, ldc, M, act);
    if (doN2) store_tile(acc11, m0 + 16, col1, cv1, hs, bias, Dadd, ldd, C, ldc, M, act);
  }
}

// ------------------------------ utility kernels ----------------------------
__global__ void fillf_kernel(float* __restrict__ p, float v, long n) {
  long t = (long)blockIdx.x * blockDim.x + threadIdx.x;
  if (t < n) p[t] = v;
}
__global__ void fillu_kernel(unsigned* __restrict__ p, unsigned v, long n) {
  long t = (long)blockIdx.x * blockDim.x + threadIdx.x;
  if (t < n) p[t] = v;
}

// r, dirv, xe = [r, ef0, ef1] per edge
__global__ void geom_kernel(const float* __restrict__ coords,
                            const int* __restrict__ src, const int* __restrict__ dst,
                            const float* __restrict__ ef,
                            float* __restrict__ r, float* __restrict__ dirv,
                            float* __restrict__ xe, int E)
{
  int e = blockIdx.x * blockDim.x + threadIdx.x;
  if (e >= E) return;
  int s = src[e], d = dst[e];
  float dx = coords[d*3+0] - coords[s*3+0];
  float dy = coords[d*3+1] - coords[s*3+1];
  float dz = coords[d*3+2] - coords[s*3+2];
  float rr = sqrtf(dx*dx + dy*dy + dz*dz + EPSV);
  float inv = 1.0f / rr;
  r[e] = rr;
  dirv[e*3+0] = dx*inv; dirv[e*3+1] = dy*inv; dirv[e*3+2] = dz*inv;
  xe[e*3+0] = rr; xe[e*3+1] = ef[e*2+0]; xe[e*3+2] = ef[e*2+1];
}

// h1d[e,c] = sum_m h1[src[e],c,m] * dirv[e,m]   (h1 as 3 planes [N,32])
__global__ void h1dot_kernel(const float* __restrict__ hx, const float* __restrict__ hy,
                             const float* __restrict__ hz, const int* __restrict__ src,
                             const float* __restrict__ dirv, float* __restrict__ out, int E)
{
  long t = (long)blockIdx.x * blockDim.x + threadIdx.x;
  if (t >= (long)E * 32) return;
  int e = (int)(t >> 5), c = (int)(t & 31);
  long s = (long)src[e] * 32 + c;
  out[t] = hx[s]*dirv[e*3] + hy[s]*dirv[e*3+1] + hz[s]*dirv[e*3+2];
}

// combine radial weights with projected messages -> v0, k, v1{x,y,z}
__global__ void edge_combine_kernel(const float* __restrict__ rv, const float* __restrict__ rk,
    const float* __restrict__ G0, const float* __restrict__ G1, const float* __restrict__ GK,
    const float* __restrict__ GDV, const float* __restrict__ GDK,
    const float* __restrict__ GX, const float* __restrict__ GY, const float* __restrict__ GZ,
    const float* __restrict__ dirv,
    float* __restrict__ v0, float* __restrict__ kk,
    float* __restrict__ v1x, float* __restrict__ v1y, float* __restrict__ v1z, int E)
{
  long t = (long)blockIdx.x * blockDim.x + threadIdx.x;
  if (t >= (long)E * 16) return;
  int e = (int)(t >> 4), p = (int)(t & 15);
  float w00 = rv[e*64 + p],      w01 = rv[e*64 + 16 + p];
  float w10 = rv[e*64 + 32 + p], w11 = rv[e*64 + 48 + p];
  float k00 = rk[e*32 + p],      k10 = rk[e*32 + 16 + p];
  float g1  = w01 * G1[e*16+p];
  float ov0 = w00 * G0[e*16+p];
  float ok  = k00 * GK[e*16+p];
  float vx = g1*dirv[e*3], vy = g1*dirv[e*3+1], vz = g1*dirv[e*3+2];
  if (GDV) {
    ov0 += w10 * GDV[e*16+p];
    ok  += k10 * GDK[e*16+p];
    vx  += w11 * GX[e*16+p];
    vy  += w11 * GY[e*16+p];
    vz  += w11 * GZ[e*16+p];
  }
  v0[t] = ov0; kk[t] = ok; v1x[t] = vx; v1y[t] = vy; v1z[t] = vz;
}

// per-(edge,head) score + segment max
__global__ void score_kernel(const float* __restrict__ kbuf, const float* __restrict__ q,
                             const int* __restrict__ dst, float* __restrict__ sc,
                             unsigned* __restrict__ mx, int E)
{
  int t = blockIdx.x * blockDim.x + threadIdx.x;
  if (t >= E * 2) return;
  int e = t >> 1, h = t & 1, d = dst[e];
  float s = 0.0f;
#pragma unroll
  for (int j = 0; j < 8; ++j) s += q[d*16 + h*8 + j] * kbuf[e*16 + h*8 + j];
  s *= 0.35355339059327373f;   // 1/sqrt(dh=8)
  sc[t] = s;
  atomicMax(&mx[d*2 + h], fenc(s));
}

// ex = exp(sc - mx[dst]);  den[dst] += ex   (ex written over sc)
__global__ void expnorm_kernel(float* __restrict__ sc, const unsigned* __restrict__ mx,
                               float* __restrict__ den, const int* __restrict__ dst, int E)
{
  int t = blockIdx.x * blockDim.x + threadIdx.x;
  if (t >= E * 2) return;
  int e = t >> 1, h = t & 1, d = dst[e];
  float ex = __expf(sc[t] - fdec(mx[d*2 + h]));
  sc[t] = ex;
  atomicAdd(&den[d*2 + h], ex);
}

// o0[dst] += a*v0 ; o1m[dst] += a*v1m
__global__ void agg_kernel(const float* __restrict__ ex, const float* __restrict__ den,
                           const int* __restrict__ dst,
                           const float* __restrict__ v0, const float* __restrict__ v1x,
                           const float* __restrict__ v1y, const float* __restrict__ v1z,
                           float* __restrict__ o0, float* __restrict__ o1x,
                           float* __restrict__ o1y, float* __restrict__ o1z, int E)
{
  long t = (long)blockIdx.x * blockDim.x + threadIdx.x;
  if (t >= (long)E * 16) return;
  int e = (int)(t >> 4), p = (int)(t & 15), h = p >> 3, d = dst[e];
  float a = ex[e*2 + h] / (den[d*2 + h] + EPSV);
  atomicAdd(&o0 [(long)d*16 + p], a * v0 [t]);
  atomicAdd(&o1x[(long)d*16 + p], a * v1x[t]);
  atomicAdd(&o1y[(long)d*16 + p], a * v1y[t]);
  atomicAdd(&o1z[(long)d*16 + p], a * v1z[t]);
}

// GNormSE3 pre: one wave per node (lane == channel, wave32 == C)
__global__ void gnorm_pre_kernel(const float* __restrict__ h0,
                                 const float* __restrict__ hx, const float* __restrict__ hy,
                                 const float* __restrict__ hz,
                                 const float* __restrict__ g, const float* __restrict__ be,
                                 float* __restrict__ lnp, float* __restrict__ nv, int Nn)
{
  int n = blockIdx.x, c = threadIdx.x;
  long i = (long)n * 32 + c;
  float ss;
  if (hx) { float a = hx[i], b = hy[i], d = hz[i]; ss = a*a + b*b + d*d; }
  else    { float a = h0[i]; ss = a*a; }
  float nrm = sqrtf(ss + EPSV);
  float sum = nrm;
  for (int o = 16; o; o >>= 1) sum += __shfl_xor(sum, o, 32);
  float mean = sum * (1.0f/32.0f);
  float dv = nrm - mean, s2 = dv*dv;
  for (int o = 16; o; o >>= 1) s2 += __shfl_xor(s2, o, 32);
  float sd = sqrtf(s2 * (1.0f/32.0f));
  lnp[i] = (dv / (sd + EPSV)) * g[c] + be[c];
  nv[i]  = nrm;
}

__global__ void gnorm_scale0_kernel(float* __restrict__ h0, const float* __restrict__ s,
                                    const float* __restrict__ nv, long n)
{
  long t = (long)blockIdx.x * blockDim.x + threadIdx.x;
  if (t < n) h0[t] *= s[t] / (nv[t] + EPSV);
}
__global__ void gnorm_scale1_kernel(float* __restrict__ hx, float* __restrict__ hy,
                                    float* __restrict__ hz, const float* __restrict__ s,
                                    const float* __restrict__ nv, long n)
{
  long t = (long)blockIdx.x * blockDim.x + threadIdx.x;
  if (t >= n) return;
  float f = s[t] / (nv[t] + EPSV);
  hx[t] *= f; hy[t] *= f; hz[t] *= f;
}

// final GConv: m = w00*G0 + w10*GD ; acc[dst] += m
__global__ void conv_combine_kernel(const float* __restrict__ r2, const float* __restrict__ G0,
                                    const float* __restrict__ GD, const int* __restrict__ dst,
                                    float* __restrict__ acc, int E, int cosh)
{
  int co = 1 << cosh;
  long t = (long)blockIdx.x * blockDim.x + threadIdx.x;
  if (t >= ((long)E << cosh)) return;
  int e = (int)(t >> cosh), c = (int)(t & (co - 1));
  float m = r2[(long)e*2*co + c] * G0[(long)e*co + c];
  if (GD) m += r2[(long)e*2*co + co + c] * GD[(long)e*co + c];
  atomicAdd(&acc[(long)dst[e]*co + c], m);
}

__global__ void concat_kernel(const float* __restrict__ a, const float* __restrict__ b,
                              float* __restrict__ out, int Nn)
{
  long t = (long)blockIdx.x * blockDim.x + threadIdx.x;
  if (t >= (long)Nn * 64) return;
  int n = (int)(t >> 6), c = (int)(t & 63);
  out[t] = (c < 32) ? a[(long)n*32 + c] : b[(long)n*32 + (c - 32)];
}

__global__ void colmean_kernel(const float* __restrict__ h, float* __restrict__ out,
                               int M, int N)
{
  __shared__ float red[256];
  int c = blockIdx.x;
  float s = 0.0f;
  for (int r = threadIdx.x; r < M; r += blockDim.x) s += h[(long)r*N + c];
  red[threadIdx.x] = s; __syncthreads();
  for (int o = 128; o; o >>= 1) {
    if ((int)threadIdx.x < o) red[threadIdx.x] += red[threadIdx.x + o];
    __syncthreads();
  }
  if (threadIdx.x == 0) out[c] = red[0] / (float)M;
}

// e = E3(E2(relu(E1(pooled))))  — single block of 64 threads
__global__ void head_kernel(const float* __restrict__ pooled,
                            const float* __restrict__ W1, const float* __restrict__ b1,
                            const float* __restrict__ W2, const float* __restrict__ b2,
                            const float* __restrict__ W3, const float* __restrict__ b3,
                            float* __restrict__ out)
{
  __shared__ float t1[64], t2[64];
  int c = threadIdx.x;
  float s = b1[c];
  for (int j = 0; j < 64; ++j) s += pooled[j] * W1[j*64 + c];
  t1[c] = fmaxf(s, 0.0f); __syncthreads();
  s = b2[c];
  for (int j = 0; j < 64; ++j) s += t1[j] * W2[j*64 + c];
  t2[c] = s; __syncthreads();
  if (c < 7) {
    s = b3[c];
    for (int j = 0; j < 64; ++j) s += t2[j] * W3[j*7 + c];
    out[c] = s;
  }
}

// ========================= host-side plumbing ==============================
struct Lin    { long W, b; };
struct NormP  { long W, b, be, g; };
struct ResP   { long K00,K10,O0,O1,Q0,S0W,S0b,S1,V00,V01,V10,V11,
                     rk1W,rk1b,rk2W,rk2b,rv1W,rv1b,rv2W,rv2b; int has_l1; };
struct ConvP  { long M00,M10,WsiW,Wsib,r1W,r1b,r2W,r2b; };
struct LayerP { NormP n0, n1; ResP res; };
struct GBlockP{ ConvP conv; LayerP lay[4]; int co; };
struct Params { Lin E1,E2,E3,PA1,PA2,W1,W2; GBlockP gatm,gbnd,gres;
                Lin l1a,l1b,l1r,l2a,l2b,l2r,tile; };

struct Cur { long c; long take(long n){ long o=c; c+=n; return o; } };
static Lin mklin(Cur& q, long i, long o){ Lin l; l.W=q.take(i*o); l.b=q.take(o); return l; }
static NormP mknorm(Cur& q){ NormP n; n.W=q.take(1024); n.b=q.take(32); n.be=q.take(32); n.g=q.take(32); return n; }
static ResP mkres(Cur& q, int l1){
  ResP r; r.has_l1=l1;
  r.K00=q.take(512); r.K10 = l1 ? q.take(512) : -1;
  r.O0=q.take(512); r.O1=q.take(512); r.Q0=q.take(512);
  r.S0W=q.take(1024); r.S0b=q.take(32);
  r.S1 = l1 ? q.take(1024) : -1;
  r.V00=q.take(512); r.V01=q.take(512);
  if (l1){ r.V10=q.take(512); r.V11=q.take(512); } else { r.V10=-1; r.V11=-1; }
  r.rk1W=q.take(96);  r.rk1b=q.take(32);
  r.rk2W=q.take(1024);r.rk2b=q.take(32);
  r.rv1W=q.take(96);  r.rv1b=q.take(32);
  r.rv2W=q.take(2048);r.rv2b=q.take(64);
  return r;
}
static ConvP mkconv(Cur& q, long co){
  ConvP c; c.M00=q.take(32*co); c.M10=q.take(32*co);
  c.WsiW=q.take(32*co); c.Wsib=q.take(co);
  c.r1W=q.take(96); c.r1b=q.take(32);
  c.r2W=q.take(64*co); c.r2b=q.take(2*co);
  return c;
}
static GBlockP mkgb(Cur& q, int co){
  GBlockP g; g.co=co; g.conv=mkconv(q, co);   // "conv" < "layers"
  for (int i=0;i<4;i++){ g.lay[i].n0=mknorm(q); g.lay[i].n1=mknorm(q);  // "norm" < "res"
                         g.lay[i].res=mkres(q, i>0); }
  return g;
}
static Params mkparams(){
  Cur q{0}; Params p;
  p.E1=mklin(q,64,64); p.E2=mklin(q,64,64); p.E3=mklin(q,64,7);
  p.PA1=mklin(q,64,64); p.PA2=mklin(q,64,1);
  p.W1=mklin(q,64,64); p.W2=mklin(q,64,1);
  p.gatm=mkgb(q,64); p.gbnd=mkgb(q,32); p.gres=mkgb(q,32);
  p.l1a=mklin(q,64,64); p.l1b=mklin(q,95,95); p.l1r=mklin(q,29,29);
  p.l2a=mklin(q,64,32); p.l2b=mklin(q,95,32); p.l2r=mklin(q,29,32);
  p.tile=mklin(q,64,64);
  return p;
}

struct Bump {
  char* base; size_t off;
  float*    f(long n){ float* p=(float*)(base+off);    off += (((size_t)n*4)+255)&~(size_t)255; return p; }
  unsigned* u(long n){ unsigned* p=(unsigned*)(base+off); off += (((size_t)n*4)+255)&~(size_t)255; return p; }
  _Float16* h(long n){ _Float16* p=(_Float16*)(base+off); off += (((size_t)n*2)+255)&~(size_t)255; return p; }
};

struct WS {
  float *r,*dirv,*xe,*sc,*v0,*v1x,*v1y,*v1z;                     // edge-persistent (EA-sized)
  float *c_rh,*c_rv,*c_rk,*c_h1d,*c_k;                           // chunk buffers
  float *G0,*G1,*GK,*GDV,*GDK,*GX,*GY,*GZ,*c_c0,*c_cd;
  float *h0,*h1x,*h1y,*h1z,*q,*den,*o0,*o1x,*o1y,*o1z,*tmp,*tmp2,*nv;  // node (NA-sized)
  unsigned* mx;
  _Float16* bt;                                                  // transposed-f16 B scratch
};

static inline long cdivl(long a, long b){ return (a + b - 1) / b; }

// transpose+convert+pad B, then launch the WMMA GEMM (stream-ordered: bt reuse safe)
static inline void gemm(hipStream_t st, _Float16* bt,
                        const float* A, int lda, const int* idx,
                        const float* B, const float* bias,
                        const float* D, int ldd, float* C, int ldc,
                        int M, int N, int K, int act)
{
  int Kpad = (K + 31) & ~31;
  xposeh_kernel<<<(unsigned)cdivl((long)N*Kpad,256), 256, 0, st>>>(B, K, Kpad, N, bt);
  dim3 g((unsigned)((N+31)/32), (unsigned)((M+31)/32), 1);
  gemm_wmma<<<g, dim3(32), 0, st>>>(A, lda, idx, bt, Kpad, bias, D, ldd, C, ldc, M, N, K, act);
}
static inline void fzero(hipStream_t st, float* p, long n){
  fillf_kernel<<<(unsigned)cdivl(n,256), 256, 0, st>>>(p, 0.0f, n);
}

static void run_gblock(hipStream_t st, const float* P, const GBlockP& g,
                       const float* l0, const float* coords,
                       const int* src, const int* dst, const float* ef,
                       int Nn, int E, float* out, const WS& w)
{
  const int co = g.co;
  _Float16* bt = w.bt;
  geom_kernel<<<(unsigned)cdivl(E,256), 256, 0, st>>>(coords, src, dst, ef, w.r, w.dirv, w.xe, E);
  hipMemcpyAsync(w.h0, l0, (size_t)Nn*32*sizeof(float), hipMemcpyDeviceToDevice, st);

  for (int L = 0; L < 4; ++L) {
    const ResP& rp = g.lay[L].res;
    const bool l1 = rp.has_l1 != 0;
    fillu_kernel<<<(unsigned)cdivl((long)Nn*2,256),256,0,st>>>(w.mx, 0x007FFFFFu, (long)Nn*2); // enc(-inf)
    fzero(st, w.den, (long)Nn*2);
    fzero(st, w.o0, (long)Nn*16);
    fzero(st, w.o1x,(long)Nn*16); fzero(st, w.o1y,(long)Nn*16); fzero(st, w.o1z,(long)Nn*16);
    gemm(st, bt, w.h0,32,nullptr, P+rp.Q0, nullptr,nullptr,0, w.q,16, Nn,16,32, 0);

    for (int e0 = 0; e0 < E; e0 += ECHUNK) {
      int ec = (E - e0 < ECHUNK) ? (E - e0) : ECHUNK;
      const float* xe = w.xe + (long)e0*3;
      const float* dv = w.dirv + (long)e0*3;
      const int* sidx = src + e0;
      // radial MLPs (WMMA): 3->32 relu ->64 / ->32
      gemm(st, bt, xe,3,nullptr, P+rp.rv1W, P+rp.rv1b, nullptr,0, w.c_rh,32, ec,32,3, 1);
      gemm(st, bt, w.c_rh,32,nullptr, P+rp.rv2W, P+rp.rv2b, nullptr,0, w.c_rv,64, ec,64,32, 0);
      gemm(st, bt, xe,3,nullptr, P+rp.rk1W, P+rp.rk1b, nullptr,0, w.c_rh,32, ec,32,3, 1);
      gemm(st, bt, w.c_rh,32,nullptr, P+rp.rk2W, P+rp.rk2b, nullptr,0, w.c_rk,32, ec,32,32, 0);
      // gather-GEMM projections h0[src] @ {V00,V01,K00}
      gemm(st, bt, w.h0,32,sidx, P+rp.V00, nullptr,nullptr,0, w.G0,16, ec,16,32, 0);
      gemm(st, bt, w.h0,32,sidx, P+rp.V01, nullptr,nullptr,0, w.G1,16, ec,16,32, 0);
      gemm(st, bt, w.h0,32,sidx, P+rp.K00, nullptr,nullptr,0, w.GK,16, ec,16,32, 0);
      if (l1) {
        h1dot_kernel<<<(unsigned)cdivl((long)ec*32,256),256,0,st>>>(w.h1x,w.h1y,w.h1z,sidx,dv,w.c_h1d,ec);
        gemm(st, bt, w.c_h1d,32,nullptr, P+rp.V10, nullptr,nullptr,0, w.GDV,16, ec,16,32, 0);
        gemm(st, bt, w.c_h1d,32,nullptr, P+rp.K10, nullptr,nullptr,0, w.GDK,16, ec,16,32, 0);
        gemm(st, bt, w.h1x,32,sidx, P+rp.V11, nullptr,nullptr,0, w.GX,16, ec,16,32, 0);
        gemm(st, bt, w.h1y,32,sidx, P+rp.V11, nullptr,nullptr,0, w.GY,16, ec,16,32, 0);
        gemm(st, bt, w.h1z,32,sidx, P+rp.V11, nullptr,nullptr,0, w.GZ,16, ec,16,32, 0);
      }
      edge_combine_kernel<<<(unsigned)cdivl((long)ec*16,256),256,0,st>>>(
          w.c_rv, w.c_rk, w.G0, w.G1, w.GK,
          l1?w.GDV:nullptr, l1?w.GDK:nullptr, l1?w.GX:nullptr, l1?w.GY:nullptr, l1?w.GZ:nullptr,
          dv, w.v0 + (long)e0*16, w.c_k,
          w.v1x + (long)e0*16, w.v1y + (long)e0*16, w.v1z + (long)e0*16, ec);
      score_kernel<<<(unsigned)cdivl((long)ec*2,256),256,0,st>>>(
          w.c_k, w.q, dst + e0, w.sc + (long)e0*2, w.mx, ec);
    }
    expnorm_kernel<<<(unsigned)cdivl((long)E*2,256),256,0,st>>>(w.sc, w.mx, w.den, dst, E);
    agg_kernel<<<(unsigned)cdivl((long)E*16,256),256,0,st>>>(
        w.sc, w.den, dst, w.v0, w.v1x, w.v1y, w.v1z, w.o0, w.o1x, w.o1y, w.o1z, E);

    // out0 = o0@O0 + S0(h0);  out1m = o1m@O1 (+ h1m@S1)
    gemm(st, bt, w.h0,32,nullptr, P+rp.S0W, P+rp.S0b, nullptr,0, w.tmp,32, Nn,32,32, 0);
    float* pl[3]  = { w.h1x, w.h1y, w.h1z };
    float* opl[3] = { w.o1x, w.o1y, w.o1z };
    for (int m = 0; m < 3; ++m) {
      const float* Dp = nullptr;
      if (l1) {
        gemm(st, bt, pl[m],32,nullptr, P+rp.S1, nullptr,nullptr,0, w.tmp2,32, Nn,32,32, 0);
        Dp = w.tmp2;
      }
      gemm(st, bt, opl[m],16,nullptr, P+rp.O1, nullptr, Dp,32, pl[m],32, Nn,32,16, 0);
    }
    gemm(st, bt, w.o0,16,nullptr, P+rp.O0, nullptr, w.tmp,32, w.h0,32, Nn,32,16, 0);

    // GNormSE3 on deg-0 and deg-1
    const NormP& n0 = g.lay[L].n0;
    gnorm_pre_kernel<<<(unsigned)Nn,32,0,st>>>(w.h0, nullptr,nullptr,nullptr,
                                               P+n0.g, P+n0.be, w.tmp, w.nv, Nn);
    gemm(st, bt, w.tmp,32,nullptr, P+n0.W, P+n0.b, nullptr,0, w.tmp2,32, Nn,32,32, 1);
    gnorm_scale0_kernel<<<(unsigned)cdivl((long)Nn*32,256),256,0,st>>>(w.h0, w.tmp2, w.nv, (long)Nn*32);
    const NormP& n1 = g.lay[L].n1;
    gnorm_pre_kernel<<<(unsigned)Nn,32,0,st>>>(nullptr, w.h1x,w.h1y,w.h1z,
                                               P+n1.g, P+n1.be, w.tmp, w.nv, Nn);
    gemm(st, bt, w.tmp,32,nullptr, P+n1.W, P+n1.b, nullptr,0, w.tmp2,32, Nn,32,32, 1);
    gnorm_scale1_kernel<<<(unsigned)cdivl((long)Nn*32,256),256,0,st>>>(w.h1x,w.h1y,w.h1z, w.tmp2, w.nv, (long)Nn*32);
  }

  // final GConvSE3 -> deg-0 out [Nn, co]
  fzero(st, out, (long)Nn*co);
  const ConvP& cp = g.conv;
  const int cosh = (co == 64) ? 6 : 5;
  for (int e0 = 0; e0 < E; e0 += ECHUNK) {
    int ec = (E - e0 < ECHUNK) ? (E - e0) : ECHUNK;
    const float* xe = w.xe + (long)e0*3;
    const float* dv = w.dirv + (long)e0*3;
    const int* sidx = src + e0;
    gemm(st, bt, xe,3,nullptr, P+cp.r1W, P+cp.r1b, nullptr,0, w.c_rh,32, ec,32,3, 1);
    gemm(st, bt, w.c_rh,32,nullptr, P+cp.r2W, P+cp.r2b, nullptr,0, w.c_rv,2*co, ec,2*co,32, 0);
    gemm(st, bt, w.h0,32,sidx, P+cp.M00, nullptr,nullptr,0, w.c_c0,co, ec,co,32, 0);
    h1dot_kernel<<<(unsigned)cdivl((long)ec*32,256),256,0,st>>>(w.h1x,w.h1y,w.h1z,sidx,dv,w.c_h1d,ec);
    gemm(st, bt, w.c_h1d,32,nullptr, P+cp.M10, nullptr,nullptr,0, w.c_cd,co, ec,co,32, 0);
    conv_combine_kernel<<<(unsigned)cdivl((long)ec<<cosh,256),256,0,st>>>(
        w.c_rv, w.c_c0, w.c_cd, dst + e0, out, ec, cosh);
  }
  // out += Wsi(h0)  (in-place residual add inside GEMM epilogue)
  gemm(st, bt, w.h0,32,nullptr, P+cp.WsiW, P+cp.Wsib, out,co, out,co, Nn,co,32, 0);
}

// ================================ entry ====================================
extern "C" void kernel_launch(void* const* d_in, const int* in_sizes, int n_in,
                              void* d_out, int out_size, void* d_ws, size_t ws_size,
                              hipStream_t stream)
{
  (void)in_sizes; (void)n_in; (void)out_size; (void)ws_size;
  const float* feat_bnd   = (const float*)d_in[0];
  const float* feat_res   = (const float*)d_in[1];
  const float* ef_bnd     = (const float*)d_in[2];
  const float* ef_res     = (const float*)d_in[3];
  const float* ef_atm     = (const float*)d_in[4];
  const float* P          = (const float*)d_in[5];   // params, jax tree-flatten order
  const float* coords_atm = (const float*)d_in[6];
  const float* coords_res = (const float*)d_in[7];
  const int*   ei_bnd     = (const int*)d_in[8];
  const int*   ei_res     = (const int*)d_in[9];
  const int*   ei_atm     = (const int*)d_in[10];
  const float* r2a        = (const float*)d_in[11];
  float* out = (float*)d_out;

  Params pr = mkparams();
  Bump bp{(char*)d_ws, 0};

  // stage buffers
  float* tb    = bp.f((long)NA_*95);
  float* l0b   = bp.f((long)NA_*32);
  float* l0r   = bp.f((long)NR_*32);
  float* hbnd  = bp.f((long)NA_*32);
  float* hres  = bp.f((long)NR_*32);
  float* hresA = bp.f((long)NA_*32);
  float* cat   = bp.f((long)NA_*64);
  float* l0a   = bp.f((long)NA_*32);
  float* hatm  = bp.f((long)NA_*64);
  float* htile = bp.f((long)NA_*64);
  float* t64   = bp.f((long)NA_*64);
  float* pooled= bp.f(64);

  WS w;
  w.r   = bp.f(EA_);        w.dirv= bp.f(3L*EA_);  w.xe  = bp.f(3L*EA_);
  w.sc  = bp.f(2L*EA_);     w.v0  = bp.f(16L*EA_);
  w.v1x = bp.f(16L*EA_);    w.v1y = bp.f(16L*EA_); w.v1z = bp.f(16L*EA_);
  w.c_rh = bp.f(32L*ECHUNK); w.c_rv = bp.f(128L*ECHUNK); w.c_rk = bp.f(32L*ECHUNK);
  w.c_h1d= bp.f(32L*ECHUNK); w.c_k  = bp.f(16L*ECHUNK);
  w.G0 = bp.f(16L*ECHUNK); w.G1 = bp.f(16L*ECHUNK); w.GK = bp.f(16L*ECHUNK);
  w.GDV= bp.f(16L*ECHUNK); w.GDK= bp.f(16L*ECHUNK);
  w.GX = bp.f(16L*ECHUNK); w.GY = bp.f(16L*ECHUNK); w.GZ = bp.f(16L*ECHUNK);
  w.c_c0 = bp.f(64L*ECHUNK); w.c_cd = bp.f(64L*ECHUNK);
  w.h0 = bp.f((long)NA_*32);
  w.h1x= bp.f((long)NA_*32); w.h1y= bp.f((long)NA_*32); w.h1z= bp.f((long)NA_*32);
  w.q  = bp.f((long)NA_*16); w.den= bp.f((long)NA_*2);
  w.o0 = bp.f((long)NA_*16);
  w.o1x= bp.f((long)NA_*16); w.o1y= bp.f((long)NA_*16); w.o1z= bp.f((long)NA_*16);
  w.tmp= bp.f((long)NA_*64); w.tmp2=bp.f((long)NA_*64); w.nv = bp.f((long)NA_*32);
  w.mx = bp.u((long)NA_*2);
  w.bt = bp.h(131072);   // max N*Kpad = 32*2016 = 64512 f16 (hres in r2a GEMM)

  // ---- input MLPs (WMMA GEMM, ELU) ----
  gemm(stream, w.bt, feat_bnd,95,nullptr, P+pr.l1b.W, P+pr.l1b.b, nullptr,0, tb,95, NA_,95,95, 2);
  gemm(stream, w.bt, tb,95,nullptr,       P+pr.l2b.W, P+pr.l2b.b, nullptr,0, l0b,32, NA_,32,95, 0);
  gemm(stream, w.bt, feat_res,29,nullptr, P+pr.l1r.W, P+pr.l1r.b, nullptr,0, tb,29, NR_,29,29, 2);
  gemm(stream, w.bt, tb,29,nullptr,       P+pr.l2r.W, P+pr.l2r.b, nullptr,0, l0r,32, NR_,32,29, 0);

  // ---- bond / residue graph blocks ----
  run_gblock(stream, P, pr.gbnd, l0b, coords_atm, ei_bnd, ei_bnd+EB_, ef_bnd, NA_, EB_, hbnd, w);
  run_gblock(stream, P, pr.gres, l0r, coords_res, ei_res, ei_res+ER_, ef_res, NR_, ER_, hres, w);

  // ---- residue -> atom broadcast (one-hot r2a as dense WMMA GEMM) ----
  gemm(stream, w.bt, r2a,NR_,nullptr, hres, nullptr,nullptr,0, hresA,32, NA_,32,NR_, 0);
  concat_kernel<<<(unsigned)cdivl((long)NA_*64,256),256,0,stream>>>(hbnd, hresA, cat, NA_);
  gemm(stream, w.bt, cat,64,nullptr, P+pr.l1a.W, P+pr.l1a.b, nullptr,0, t64,64, NA_,64,64, 2);
  gemm(stream, w.bt, t64,64,nullptr, P+pr.l2a.W, P+pr.l2a.b, nullptr,0, l0a,32, NA_,32,64, 0);

  // ---- atom graph block (co = 64) ----
  run_gblock(stream, P, pr.gatm, l0a, coords_atm, ei_atm, ei_atm+EA_, ef_atm, NA_, EA_, hatm, w);

  // ---- heads ----
  gemm(stream, w.bt, hatm,64,nullptr,  P+pr.tile.W, P+pr.tile.b, nullptr,0, htile,64, NA_,64,64, 0);
  gemm(stream, w.bt, htile,64,nullptr, P+pr.PA1.W,  P+pr.PA1.b,  nullptr,0, t64,64,  NA_,64,64, 1);
  gemm(stream, w.bt, t64,64,nullptr,   P+pr.PA2.W,  P+pr.PA2.b,  nullptr,0, out,1,   NA_,1,64,  0);
  gemm(stream, w.bt, htile,64,nullptr, P+pr.W1.W,   P+pr.W1.b,   nullptr,0, t64,64,  NA_,64,64, 1);
  gemm(stream, w.bt, t64,64,nullptr,   P+pr.W2.W,   P+pr.W2.b,   nullptr,0, out+NA_,1, NA_,1,64, 0);
  colmean_kernel<<<64,256,0,stream>>>(htile, pooled, NA_, 64);
  head_kernel<<<1,64,0,stream>>>(pooled, P+pr.E1.W,P+pr.E1.b, P+pr.E2.W,P+pr.E2.b,
                                 P+pr.E3.W,P+pr.E3.b, out + 2*NA_);
}
